// FNetEncoder_67826123538564
// MI455X (gfx1250) — matrix operand order repaired and checked by reading, not compiled, not run on hardware
//
#include <hip/hip_runtime.h>
#include <hip/hip_bf16.h>

typedef __attribute__((ext_vector_type(16))) __bf16 v16bf;
typedef __attribute__((ext_vector_type(8)))  float  v8f;

#define BM 128
#define BN 64
#define BK 32
#define LDA_P 40   // 32 + 8 pad (bf16 units) -> 80B rows, 16B-aligned
#define LDB_P 72   // 64 + 8 pad (bf16 units) -> 144B rows, 16B-aligned

__device__ __forceinline__ __bf16 f2bf(float f) { return (__bf16)f; }

// Low 32 bits of a generic pointer to __shared__ == LDS byte offset
// (local->generic addrspacecast is {aperture_hi, lds_offset}; async-to-LDS
//  adds the VDST VGPR to the LDS base, ISA 10.2 / 15.18.3).
__device__ __forceinline__ unsigned lds_off(const void* p) {
  return (unsigned)(unsigned long long)p;
}

// 16B global -> LDS async copy (VGLOBAL op 98, ASYNCcnt-tracked)
#define ASYNC16(ldsoff, gptr)                                             \
  asm volatile("global_load_async_to_lds_b128 %0, %1, off"                \
               :: "v"(ldsoff), "v"(gptr) : "memory")

#define WAIT_ASYNC(n)                                                     \
  asm volatile("s_wait_asynccnt " #n ::: "memory")

// ---- WMMA fragment gathers (ISA 7.12.2 layouts, wave32) --------------------
__device__ __forceinline__ v16bf frag_a(const __bf16* s, int m0, int lane) {
  const int m  = m0 + (lane & 15);
  const int kh = lane >> 4;
  v16bf a;
#pragma unroll
  for (int v = 0; v < 8; ++v) {
    const int k0 = ((v & 4) << 2) + kh * 8 + (v & 3) * 2;
    a[2 * v]     = s[m * LDA_P + k0];
    a[2 * v + 1] = s[m * LDA_P + k0 + 1];
  }
  return a;
}

__device__ __forceinline__ v16bf frag_b(const __bf16* s, int n0, int lane) {
  v16bf b;
#pragma unroll
  for (int v = 0; v < 16; ++v) b[v] = s[(((v & 8) << 1) + (lane >> 4) * 8 + (v & 7)) * LDB_P + n0 + (lane & 15)];
  return b;
}

#define WMMA(acc, a, b) \
  acc = __builtin_amdgcn_wmma_f32_16x16x32_bf16(false, (a), false, (b), (short)0, (acc), false, false)

// ---- DFT matrix generation: C[i,j]=cos(2*pi*i*j/n), S[i,j]=sin(...) --------
__global__ void k_gendft(__bf16* __restrict__ c, __bf16* __restrict__ s, int n) {
  const long i = (long)blockIdx.x * blockDim.x + threadIdx.x;
  const long total = (long)n * n;
  if (i >= total) return;
  const int r = (int)(i / n), q = (int)(i % n);
  const int p = (int)(((long)r * q) % n);
  const float ang = 6.283185307179586f * ((float)p / (float)n);
  c[i] = f2bf(cosf(ang));
  s[i] = f2bf(sinf(ang));
}

__global__ void k_cvt(const float* __restrict__ in, __bf16* __restrict__ out, long n) {
  const long i = (long)blockIdx.x * blockDim.x + threadIdx.x;
  if (i < n) out[i] = f2bf(in[i]);
}

// ---- LayerNorm over D=512, one 128-thread block per row, bf16 out ----------
__global__ __launch_bounds__(128) void k_ln(const float* __restrict__ x,
                                            const float* __restrict__ g,
                                            const float* __restrict__ b,
                                            __bf16* __restrict__ out) {
  __shared__ float red[128];
  const int tid = threadIdx.x;
  const size_t row = (size_t)blockIdx.x * 512;
  const float v0 = x[row + tid];
  const float v1 = x[row + tid + 128];
  const float v2 = x[row + tid + 256];
  const float v3 = x[row + tid + 384];
  red[tid] = v0 + v1 + v2 + v3;
  __syncthreads();
  for (int o = 64; o > 0; o >>= 1) { if (tid < o) red[tid] += red[tid + o]; __syncthreads(); }
  const float mean = red[0] * (1.0f / 512.0f);
  __syncthreads();
  const float d0 = v0 - mean, d1 = v1 - mean, d2 = v2 - mean, d3 = v3 - mean;
  red[tid] = d0 * d0 + d1 * d1 + d2 * d2 + d3 * d3;
  __syncthreads();
  for (int o = 64; o > 0; o >>= 1) { if (tid < o) red[tid] += red[tid + o]; __syncthreads(); }
  const float rstd = rsqrtf(red[0] * (1.0f / 512.0f) + 1e-5f);
  out[row + tid]       = f2bf(d0 * rstd * g[tid]       + b[tid]);
  out[row + tid + 128] = f2bf(d1 * rstd * g[tid + 128] + b[tid + 128]);
  out[row + tid + 256] = f2bf(d2 * rstd * g[tid + 256] + b[tid + 256]);
  out[row + tid + 384] = f2bf(d3 * rstd * g[tid + 384] + b[tid + 384]);
}

// ---- DFT over D: Yc = A*CD, Ys = A*SD (shared A tile, dual B chains) -------
__global__ __launch_bounds__(256) void k_dftd(const __bf16* __restrict__ A,
                                              const __bf16* __restrict__ Bc,
                                              const __bf16* __restrict__ Bs,
                                              __bf16* __restrict__ Yc,
                                              __bf16* __restrict__ Ys,
                                              int N, int K) {
  __shared__ __bf16 sA[2][BM * LDA_P];
  __shared__ __bf16 sBc[2][BK * LDB_P];
  __shared__ __bf16 sBs[2][BK * LDB_P];
  const int tid = threadIdx.x, lane = tid & 31, wave = tid >> 5;
  const int wm = (wave & 3) * 32, wn = (wave >> 2) * 32;
  const int bm = blockIdx.y * BM, bn = blockIdx.x * BN;
  // per-thread copy slots: A row + 32B, B row + 16B
  const int ra = tid >> 1, ca = (tid & 1) * 16;        // A: 128 rows x 2 half-rows
  const int rb = tid >> 3, cb = (tid & 7) * 8;         // B: 32 rows x 8 chunks
  const __bf16* gA  = A  + (size_t)(bm + ra) * K + ca;
  const __bf16* gBc = Bc + (size_t)rb * N + bn + cb;
  const __bf16* gBs = Bs + (size_t)rb * N + bn + cb;
  unsigned oA[2], oBc[2], oBs[2];
#pragma unroll
  for (int s = 0; s < 2; ++s) {
    oA[s]  = lds_off(&sA[s][ra * LDA_P + ca]);
    oBc[s] = lds_off(&sBc[s][rb * LDB_P + cb]);
    oBs[s] = lds_off(&sBs[s][rb * LDB_P + cb]);
  }
  v8f ac[2][2] = {}, as_[2][2] = {};
  int cur = 0;
  // prefetch tile 0: 4 async instructions per wave
  ASYNC16(oA[0], gA);  ASYNC16(oA[0] + 16, gA + 8);
  ASYNC16(oBc[0], gBc); ASYNC16(oBs[0], gBs);
  for (int kk = 0; kk < K; kk += BK) {
    const bool more = (kk + BK) < K;
    if (more) {
      const int nx = cur ^ 1;
      const __bf16* a = gA + (kk + BK);
      ASYNC16(oA[nx], a);  ASYNC16(oA[nx] + 16, a + 8);
      ASYNC16(oBc[nx], gBc + (size_t)(kk + BK) * N);
      ASYNC16(oBs[nx], gBs + (size_t)(kk + BK) * N);
      WAIT_ASYNC(4);
    } else {
      WAIT_ASYNC(0);
    }
    __syncthreads();
    v16bf a0 = frag_a(sA[cur], wm, lane), a1 = frag_a(sA[cur], wm + 16, lane);
    v16bf bc0 = frag_b(sBc[cur], wn, lane), bc1 = frag_b(sBc[cur], wn + 16, lane);
    v16bf bs0 = frag_b(sBs[cur], wn, lane), bs1 = frag_b(sBs[cur], wn + 16, lane);
    WMMA(ac[0][0], a0, bc0);  WMMA(ac[0][1], a0, bc1);
    WMMA(ac[1][0], a1, bc0);  WMMA(ac[1][1], a1, bc1);
    WMMA(as_[0][0], a0, bs0); WMMA(as_[0][1], a0, bs1);
    WMMA(as_[1][0], a1, bs0); WMMA(as_[1][1], a1, bs1);
    __syncthreads();
    cur ^= 1;
  }
  const int half = lane >> 4, nl = lane & 15;
#pragma unroll
  for (int j = 0; j < 2; ++j) {
    const int n = bn + wn + j * 16 + nl;
#pragma unroll
    for (int i = 0; i < 2; ++i) {
#pragma unroll
      for (int v = 0; v < 8; ++v) {
        const size_t idx = (size_t)(bm + wm + i * 16 + v + 8 * half) * N + n;
        Yc[idx] = f2bf(ac[i][j][v]);
        Ys[idx] = f2bf(as_[i][j][v]);
      }
    }
  }
}

// ---- DFT over S: out = CS*Yc - SS*Ys + residual (per batch, f32 out) -------
__global__ __launch_bounds__(256) void k_dfts(const __bf16* __restrict__ Ac,
                                              const __bf16* __restrict__ As,
                                              const __bf16* __restrict__ Bc0,
                                              const __bf16* __restrict__ Bs0,
                                              const float* __restrict__ res0,
                                              float* __restrict__ out0,
                                              int M, int N, int K) {
  __shared__ __bf16 sAc[2][BM * LDA_P];
  __shared__ __bf16 sAs[2][BM * LDA_P];
  __shared__ __bf16 sBc[2][BK * LDB_P];
  __shared__ __bf16 sBs[2][BK * LDB_P];
  const size_t boff = (size_t)blockIdx.z * M * N;
  const __bf16* Bc = Bc0 + boff;
  const __bf16* Bs = Bs0 + boff;
  const float* res = res0 + boff;
  float* out = out0 + boff;
  const int tid = threadIdx.x, lane = tid & 31, wave = tid >> 5;
  const int wm = (wave & 3) * 32, wn = (wave >> 2) * 32;
  const int bm = blockIdx.y * BM, bn = blockIdx.x * BN;
  const int ra = tid >> 1, ca = (tid & 1) * 16;
  const int rb = tid >> 3, cb = (tid & 7) * 8;
  const __bf16* gAc = Ac + (size_t)(bm + ra) * K + ca;
  const __bf16* gAs = As + (size_t)(bm + ra) * K + ca;
  const __bf16* gBc = Bc + (size_t)rb * N + bn + cb;
  const __bf16* gBs = Bs + (size_t)rb * N + bn + cb;
  unsigned oAc[2], oAs[2], oBc[2], oBs[2];
#pragma unroll
  for (int s = 0; s < 2; ++s) {
    oAc[s] = lds_off(&sAc[s][ra * LDA_P + ca]);
    oAs[s] = lds_off(&sAs[s][ra * LDA_P + ca]);
    oBc[s] = lds_off(&sBc[s][rb * LDB_P + cb]);
    oBs[s] = lds_off(&sBs[s][rb * LDB_P + cb]);
  }
  v8f ac[2][2] = {}, as_[2][2] = {};
  int cur = 0;
  // prefetch tile 0: 6 async instructions per wave
  ASYNC16(oAc[0], gAc); ASYNC16(oAc[0] + 16, gAc + 8);
  ASYNC16(oAs[0], gAs); ASYNC16(oAs[0] + 16, gAs + 8);
  ASYNC16(oBc[0], gBc); ASYNC16(oBs[0], gBs);
  for (int kk = 0; kk < K; kk += BK) {
    const bool more = (kk + BK) < K;
    if (more) {
      const int nx = cur ^ 1;
      const __bf16* pc = gAc + (kk + BK);
      const __bf16* ps = gAs + (kk + BK);
      ASYNC16(oAc[nx], pc); ASYNC16(oAc[nx] + 16, pc + 8);
      ASYNC16(oAs[nx], ps); ASYNC16(oAs[nx] + 16, ps + 8);
      ASYNC16(oBc[nx], gBc + (size_t)(kk + BK) * N);
      ASYNC16(oBs[nx], gBs + (size_t)(kk + BK) * N);
      WAIT_ASYNC(6);
    } else {
      WAIT_ASYNC(0);
    }
    __syncthreads();
    v16bf ac0 = frag_a(sAc[cur], wm, lane), ac1 = frag_a(sAc[cur], wm + 16, lane);
    v16bf as0 = frag_a(sAs[cur], wm, lane), as1 = frag_a(sAs[cur], wm + 16, lane);
    v16bf bc0 = frag_b(sBc[cur], wn, lane), bc1 = frag_b(sBc[cur], wn + 16, lane);
    v16bf bs0 = frag_b(sBs[cur], wn, lane), bs1 = frag_b(sBs[cur], wn + 16, lane);
    WMMA(ac[0][0], ac0, bc0);  WMMA(ac[0][1], ac0, bc1);
    WMMA(ac[1][0], ac1, bc0);  WMMA(ac[1][1], ac1, bc1);
    WMMA(as_[0][0], as0, bs0); WMMA(as_[0][1], as0, bs1);
    WMMA(as_[1][0], as1, bs0); WMMA(as_[1][1], as1, bs1);
    __syncthreads();
    cur ^= 1;
  }
  const int half = lane >> 4, nl = lane & 15;
#pragma unroll
  for (int j = 0; j < 2; ++j) {
    const int n = bn + wn + j * 16 + nl;
#pragma unroll
    for (int i = 0; i < 2; ++i) {
#pragma unroll
      for (int v = 0; v < 8; ++v) {
        const size_t idx = (size_t)(bm + wm + i * 16 + v + 8 * half) * N + n;
        out[idx] = ac[i][j][v] - as_[i][j][v] + res[idx];
      }
    }
  }
}

// ---- FFN1: C = gelu(A*W1 + b1), bf16 out -----------------------------------
__global__ __launch_bounds__(256) void k_ffn1(const __bf16* __restrict__ A,
                                              const __bf16* __restrict__ B,
                                              const float* __restrict__ bias,
                                              __bf16* __restrict__ C,
                                              int N, int K) {
  __shared__ __bf16 sA[2][BM * LDA_P];
  __shared__ __bf16 sB[2][BK * LDB_P];
  const int tid = threadIdx.x, lane = tid & 31, wave = tid >> 5;
  const int wm = (wave & 3) * 32, wn = (wave >> 2) * 32;
  const int bm = blockIdx.y * BM, bn = blockIdx.x * BN;
  const int ra = tid >> 1, ca = (tid & 1) * 16;
  const int rb = tid >> 3, cb = (tid & 7) * 8;
  const __bf16* gA = A + (size_t)(bm + ra) * K + ca;
  const __bf16* gB = B + (size_t)rb * N + bn + cb;
  unsigned oA[2], oB[2];
#pragma unroll
  for (int s = 0; s < 2; ++s) {
    oA[s] = lds_off(&sA[s][ra * LDA_P + ca]);
    oB[s] = lds_off(&sB[s][rb * LDB_P + cb]);
  }
  v8f acc[2][2] = {};
  int cur = 0;
  ASYNC16(oA[0], gA); ASYNC16(oA[0] + 16, gA + 8); ASYNC16(oB[0], gB);
  for (int kk = 0; kk < K; kk += BK) {
    const bool more = (kk + BK) < K;
    if (more) {
      const int nx = cur ^ 1;
      const __bf16* a = gA + (kk + BK);
      ASYNC16(oA[nx], a); ASYNC16(oA[nx] + 16, a + 8);
      ASYNC16(oB[nx], gB + (size_t)(kk + BK) * N);
      WAIT_ASYNC(3);
    } else {
      WAIT_ASYNC(0);
    }
    __syncthreads();
    v16bf a0 = frag_a(sA[cur], wm, lane), a1 = frag_a(sA[cur], wm + 16, lane);
    v16bf b0 = frag_b(sB[cur], wn, lane), b1 = frag_b(sB[cur], wn + 16, lane);
    WMMA(acc[0][0], a0, b0); WMMA(acc[0][1], a0, b1);
    WMMA(acc[1][0], a1, b0); WMMA(acc[1][1], a1, b1);
    __syncthreads();
    cur ^= 1;
  }
  const int half = lane >> 4, nl = lane & 15;
#pragma unroll
  for (int j = 0; j < 2; ++j) {
    const int n = bn + wn + j * 16 + nl;
    const float bv = bias[n];
#pragma unroll
    for (int i = 0; i < 2; ++i) {
#pragma unroll
      for (int v = 0; v < 8; ++v) {
        const int m = bm + wm + i * 16 + v + 8 * half;
        float t = acc[i][j][v] + bv;
        t = 0.5f * t * (1.0f + erff(t * 0.70710678118654752f));  // exact gelu
        C[(size_t)m * N + n] = f2bf(t);
      }
    }
  }
}

// ---- FFN2: out = A*W2 + b2 + residual, f32 out -----------------------------
__global__ __launch_bounds__(256) void k_ffn2(const __bf16* __restrict__ A,
                                              const __bf16* __restrict__ B,
                                              const float* __restrict__ bias,
                                              const float* __restrict__ res,
                                              float* __restrict__ out,
                                              int N, int K) {
  __shared__ __bf16 sA[2][BM * LDA_P];
  __shared__ __bf16 sB[2][BK * LDB_P];
  const int tid = threadIdx.x, lane = tid & 31, wave = tid >> 5;
  const int wm = (wave & 3) * 32, wn = (wave >> 2) * 32;
  const int bm = blockIdx.y * BM, bn = blockIdx.x * BN;
  const int ra = tid >> 1, ca = (tid & 1) * 16;
  const int rb = tid >> 3, cb = (tid & 7) * 8;
  const __bf16* gA = A + (size_t)(bm + ra) * K + ca;
  const __bf16* gB = B + (size_t)rb * N + bn + cb;
  unsigned oA[2], oB[2];
#pragma unroll
  for (int s = 0; s < 2; ++s) {
    oA[s] = lds_off(&sA[s][ra * LDA_P + ca]);
    oB[s] = lds_off(&sB[s][rb * LDB_P + cb]);
  }
  v8f acc[2][2] = {};
  int cur = 0;
  ASYNC16(oA[0], gA); ASYNC16(oA[0] + 16, gA + 8); ASYNC16(oB[0], gB);
  for (int kk = 0; kk < K; kk += BK) {
    const bool more = (kk + BK) < K;
    if (more) {
      const int nx = cur ^ 1;
      const __bf16* a = gA + (kk + BK);
      ASYNC16(oA[nx], a); ASYNC16(oA[nx] + 16, a + 8);
      ASYNC16(oB[nx], gB + (size_t)(kk + BK) * N);
      WAIT_ASYNC(3);
    } else {
      WAIT_ASYNC(0);
    }
    __syncthreads();
    v16bf a0 = frag_a(sA[cur], wm, lane), a1 = frag_a(sA[cur], wm + 16, lane);
    v16bf b0 = frag_b(sB[cur], wn, lane), b1 = frag_b(sB[cur], wn + 16, lane);
    WMMA(acc[0][0], a0, b0); WMMA(acc[0][1], a0, b1);
    WMMA(acc[1][0], a1, b0); WMMA(acc[1][1], a1, b1);
    __syncthreads();
    cur ^= 1;
  }
  const int half = lane >> 4, nl = lane & 15;
#pragma unroll
  for (int j = 0; j < 2; ++j) {
    const int n = bn + wn + j * 16 + nl;
    const float bv = bias[n];
#pragma unroll
    for (int i = 0; i < 2; ++i) {
#pragma unroll
      for (int v = 0; v < 8; ++v) {
        const size_t idx = (size_t)(bm + wm + i * 16 + v + 8 * half) * N + n;
        out[idx] = acc[i][j][v] + bv + res[idx];
      }
    }
  }
}

// ---------------------------------------------------------------------------
extern "C" void kernel_launch(void* const* d_in, const int* in_sizes, int n_in,
                              void* d_out, int out_size, void* d_ws, size_t ws_size,
                              hipStream_t stream) {
  const float* x    = (const float*)d_in[0];
  const float* ln1g = (const float*)d_in[1];
  const float* ln1b = (const float*)d_in[2];
  const float* ln2g = (const float*)d_in[3];
  const float* ln2b = (const float*)d_in[4];
  const float* W1   = (const float*)d_in[5];
  const float* b1   = (const float*)d_in[6];
  const float* W2   = (const float*)d_in[7];
  const float* b2   = (const float*)d_in[8];

  const int Bb = 8, S = 4096, D = 512, H = 2048, L = 4;
  const size_t MR = (size_t)Bb * S;  // 32768 rows

  char* ws = (char*)d_ws;
  size_t off = 0;
  auto take = [&](size_t bytes) -> void* {
    void* p = ws + off;
    off += (bytes + 255) & ~(size_t)255;
    return p;
  };
  __bf16* CS  = (__bf16*)take((size_t)S * S * 2);       // 32 MB
  __bf16* SS  = (__bf16*)take((size_t)S * S * 2);       // 32 MB
  __bf16* CD  = (__bf16*)take((size_t)D * D * 2);
  __bf16* SD  = (__bf16*)take((size_t)D * D * 2);
  __bf16* W1b = (__bf16*)take((size_t)L * D * H * 2);
  __bf16* W2b = (__bf16*)take((size_t)L * H * D * 2);
  __bf16* hn  = (__bf16*)take(MR * D * 2);
  __bf16* Yc  = (__bf16*)take(MR * D * 2);
  __bf16* Ys  = (__bf16*)take(MR * D * 2);
  __bf16* h1  = (__bf16*)take(MR * H * 2);              // 128 MB

  float* xb = (float*)d_out;  // running residual buffer (f32), ends as output
  hipMemcpyAsync(xb, x, MR * D * sizeof(float), hipMemcpyDeviceToDevice, stream);

  {
    long t = (long)D * D;
    k_gendft<<<(unsigned)((t + 255) / 256), 256, 0, stream>>>(CD, SD, D);
  }
  {
    long t = (long)S * S;
    k_gendft<<<(unsigned)((t + 255) / 256), 256, 0, stream>>>(CS, SS, S);
  }
  {
    long t = (long)L * D * H;
    k_cvt<<<(unsigned)((t + 255) / 256), 256, 0, stream>>>(W1, W1b, t);
    k_cvt<<<(unsigned)((t + 255) / 256), 256, 0, stream>>>(W2, W2b, t);
  }

  for (int l = 0; l < L; ++l) {
    k_ln<<<(unsigned)MR, 128, 0, stream>>>(xb, ln1g + l * D, ln1b + l * D, hn);
    {
      dim3 g(D / BN, (unsigned)(MR / BM));
      k_dftd<<<g, 256, 0, stream>>>(hn, CD, SD, Yc, Ys, D, D);
    }
    {
      dim3 g(D / BN, S / BM, Bb);
      k_dfts<<<g, 256, 0, stream>>>(CS, SS, Yc, Ys, xb, xb, S, D, S);
    }
    k_ln<<<(unsigned)MR, 128, 0, stream>>>(xb, ln2g + l * D, ln2b + l * D, hn);
    {
      dim3 g(H / BN, (unsigned)(MR / BM));
      k_ffn1<<<g, 256, 0, stream>>>(hn, W1b + (size_t)l * D * H, b1 + (size_t)l * H, h1, H, D);
    }
    {
      dim3 g(D / BN, (unsigned)(MR / BM));
      k_ffn2<<<g, 256, 0, stream>>>(h1, W2b + (size_t)l * H * D, b2 + (size_t)l * D, xb, xb, D, H);
    }
  }
}
